// BcosGCNLayer_36429912604727
// MI455X (gfx1250) — matrix operand order repaired and checked by reading, not compiled, not run on hardware
//
#include <hip/hip_runtime.h>
#include <hip/hip_bf16.h>
#include <math.h>

// ---------------------------------------------------------------------------
// out[n,o] = lin * |lin| * (1/max(||z_n||,eps)) * (1/max(||w_o||,eps))
// lin = z @ W^T  via v_wmma_f32_16x16x32_f16.
// Fused single pass over z; z row norms accumulated during A staging;
// W row norms precomputed (2 KB workspace).
// Register-staged double-buffered LDS pipeline: one barrier per K-step,
// chunk k+1 global loads overlap chunk k WMMAs.
// ---------------------------------------------------------------------------

typedef __attribute__((ext_vector_type(16))) _Float16 v16h;
typedef __attribute__((ext_vector_type(8)))  _Float16 v8h;
typedef __attribute__((ext_vector_type(4)))  _Float16 v4h;
typedef __attribute__((ext_vector_type(8)))  float    v8f;

#define KDIM   512
#define NOUT   512
#define BM     128
#define BN     128
#define BK     32
#define APITCH 40            // halves per A row  (32 + 8 pad)  -> 80 B, odd*16B
#define BPITCH 136           // halves per B k-row(128 + 8 pad) -> 272 B, odd*16B
#define ASZ    (BM * APITCH) // one A buffer, in halves
#define BSZ    (BK * BPITCH) // one B buffer, in halves
#define EPS    1e-12f

// ---- tiny kernel: per-row inverse L2 norm of W (512 rows x 512) ------------
__global__ __launch_bounds__(256) void bcos_wnorm(const float* __restrict__ w,
                                                  float* __restrict__ iwn) {
    __shared__ float s[256];
    const int r = blockIdx.x;
    const int t = threadIdx.x;
    float x0 = w[(size_t)r * KDIM + t];
    float x1 = w[(size_t)r * KDIM + 256 + t];
    s[t] = x0 * x0 + x1 * x1;
    __syncthreads();
    for (int o = 128; o > 0; o >>= 1) {
        if (t < o) s[t] += s[t + o];
        __syncthreads();
    }
    if (t == 0) iwn[r] = 1.0f / fmaxf(sqrtf(s[0]), EPS);
}

// ---- fused GEMM + norm + Bcos scale ---------------------------------------
__global__ __launch_bounds__(256) void bcos_gemm(const float* __restrict__ z,
                                                 const float* __restrict__ w,
                                                 const float* __restrict__ iwn_g,
                                                 float* __restrict__ out,
                                                 int M) {
    __shared__ __align__(16) _Float16 As[2 * ASZ];  // row-major (m,k) f16, 2 bufs
    __shared__ __align__(16) _Float16 Bs[2 * BSZ];  // K-major  (k,n) f16, 2 bufs
    __shared__ float izn_s[BM];
    __shared__ float iwn_s[BN];

    const int tid   = threadIdx.x;
    const int lane  = tid & 31;
    const int wid   = tid >> 5;
    const int waveM = wid & 3;   // 4 strips of 32 rows
    const int waveN = wid >> 2;  // 2 strips of 64 cols
    const int hsel  = lane >> 4; // A-frag K-group select (lanes 16-31)
    const int lr    = lane & 15;

    const int n0 = blockIdx.x * BN;   // col block fastest-varying -> L2 reuse of z
    const int m0 = blockIdx.y * BM;

    if (tid < BN) iwn_s[tid] = iwn_g[n0 + tid];

    v8f acc[2][4];
    v8f zero8 = {};
#pragma unroll
    for (int i = 0; i < 2; ++i)
#pragma unroll
        for (int j = 0; j < 4; ++j) acc[i][j] = zero8;

    float ssA[4] = {0.f, 0.f, 0.f, 0.f};

    // ---- A loader mapping: thread covers 4 rows (kq fixed), clamped rows ---
    const int kq = tid & 7;                 // float4 slot within 32-wide chunk
    const float* zrow[4];
#pragma unroll
    for (int j = 0; j < 4; ++j) {
        int m  = (tid >> 3) + 32 * j;       // local row
        int gm = m0 + m;
        if (gm > M - 1) gm = M - 1;         // clamp: branchless, row data unused
        zrow[j] = z + (size_t)gm * KDIM + kq * 4;
    }
    // ---- B loader mapping: lane = k (coalesced), 16-column group -----------
    const int kk = tid & 31;
    const int ng = tid >> 5;
    const float* wcol = w + (size_t)(n0 + ng * 16) * KDIM + kk;

    float4 areg[4];
    float  breg[16];

    // ---------------- prologue: stage chunk 0 ------------------------------
#pragma unroll
    for (int j = 0; j < 4; ++j) areg[j] = *(const float4*)(zrow[j]);
#pragma unroll
    for (int i = 0; i < 16; ++i) breg[i] = wcol[(size_t)i * KDIM];

    {   // store chunk 0 into buffer 0
        _Float16* Ab = As;
        _Float16* Bb = Bs;
#pragma unroll
        for (int j = 0; j < 4; ++j) {
            float4 v = areg[j];
            ssA[j] += v.x * v.x + v.y * v.y + v.z * v.z + v.w * v.w;
            v4h h;
            h.x = (_Float16)v.x; h.y = (_Float16)v.y;
            h.z = (_Float16)v.z; h.w = (_Float16)v.w;
            *(v4h*)(Ab + ((tid >> 3) + 32 * j) * APITCH + kq * 4) = h;
        }
        v8h p0, p1;
#pragma unroll
        for (int i = 0; i < 8; ++i) p0[i] = (_Float16)breg[i];
#pragma unroll
        for (int i = 0; i < 8; ++i) p1[i] = (_Float16)breg[i + 8];
        *(v8h*)(Bb + kk * BPITCH + ng * 16)     = p0;
        *(v8h*)(Bb + kk * BPITCH + ng * 16 + 8) = p1;
    }

    // ---------------- main pipeline: 1 barrier / K-step --------------------
    int p = 0;
    for (int kb = 0; kb < KDIM; kb += BK) {
        __syncthreads();                        // buffer p ready for everyone
        const bool has_next = (kb + BK) < KDIM;

        // issue global loads for chunk k+1 (overlap with WMMAs below)
        if (has_next) {
#pragma unroll
            for (int j = 0; j < 4; ++j)
                areg[j] = *(const float4*)(zrow[j] + kb + BK);
#pragma unroll
            for (int i = 0; i < 16; ++i)
                breg[i] = wcol[(size_t)i * KDIM + kb + BK];
        }

        // fragments from buffer p + 8 WMMAs
        const _Float16* Ab = As + p * ASZ;
        const _Float16* Bb = Bs + p * BSZ;
        v16h a[2], b[4];
#pragma unroll
        for (int i = 0; i < 2; ++i) {
            // A: lane = M row; lanes<16: K {0..7,16..23}, lanes>=16: {8..15,24..31}
            const _Float16* ap = Ab + (waveM * 32 + i * 16 + lr) * APITCH + hsel * 8;
            v8h lo = *(const v8h*)ap;
            v8h hi = *(const v8h*)(ap + 16);
            a[i] = __builtin_shufflevector(lo, hi, 0, 1, 2, 3, 4, 5, 6, 7,
                                           8, 9, 10, 11, 12, 13, 14, 15);
        }
#pragma unroll
        for (int j = 0; j < 4; ++j) {
            // B: lane = N col; lanes<16: K {0..7,16..23}, lanes>=16: {8..15,24..31}
            const _Float16* bp = Bb + hsel * 8 * BPITCH + waveN * 64 + j * 16 + lr;
#pragma unroll
            for (int s = 0; s < 8; ++s) {
                b[j][s]     = bp[s * BPITCH];
                b[j][s + 8] = bp[(16 + s) * BPITCH];
            }
        }
#pragma unroll
        for (int i = 0; i < 2; ++i)
#pragma unroll
            for (int j = 0; j < 4; ++j)
                acc[i][j] = __builtin_amdgcn_wmma_f32_16x16x32_f16(
                    false, a[i], false, b[j], (short)0, acc[i][j], false, false);

        // convert + store staged chunk into the other buffer
        if (has_next) {
            _Float16* An = As + (1 - p) * ASZ;
            _Float16* Bn = Bs + (1 - p) * BSZ;
#pragma unroll
            for (int j = 0; j < 4; ++j) {
                float4 v = areg[j];
                ssA[j] += v.x * v.x + v.y * v.y + v.z * v.z + v.w * v.w;
                v4h h;
                h.x = (_Float16)v.x; h.y = (_Float16)v.y;
                h.z = (_Float16)v.z; h.w = (_Float16)v.w;
                *(v4h*)(An + ((tid >> 3) + 32 * j) * APITCH + kq * 4) = h;
            }
            v8h p0, p1;
#pragma unroll
            for (int i = 0; i < 8; ++i) p0[i] = (_Float16)breg[i];
#pragma unroll
            for (int i = 0; i < 8; ++i) p1[i] = (_Float16)breg[i + 8];
            *(v8h*)(Bn + kk * BPITCH + ng * 16)     = p0;
            *(v8h*)(Bn + kk * BPITCH + ng * 16 + 8) = p1;
        }
        p ^= 1;
    }

    // ---- reduce z row sumsq -> izn_s (reuse As LDS as float[128][8]) ------
    __syncthreads();
    {
        float* red = (float*)As;
#pragma unroll
        for (int j = 0; j < 4; ++j)
            red[((tid >> 3) + 32 * j) * 8 + (tid & 7)] = ssA[j];
        __syncthreads();
        if (tid < BM) {
            float s = 0.f;
#pragma unroll
            for (int q = 0; q < 8; ++q) s += red[tid * 8 + q];
            izn_s[tid] = 1.0f / fmaxf(sqrtf(s), EPS);
        }
        __syncthreads();
    }

    // ---- epilogue: out = lin * |lin| * izn * iwn --------------------------
    // C/D layout: VGPR vv, lane: M = vv + 8*(lane>=16), N = lane&15
#pragma unroll
    for (int i = 0; i < 2; ++i) {
        int rl_base = waveM * 32 + i * 16 + hsel * 8;
#pragma unroll
        for (int j = 0; j < 4; ++j) {
            int cl = waveN * 64 + j * 16 + lr;
            int gc = n0 + cl;
            float iw = iwn_s[cl];
#pragma unroll
            for (int vv = 0; vv < 8; ++vv) {
                int rl = rl_base + vv;
                int gr = m0 + rl;
                if (gr < M) {
                    float lin = acc[i][j][vv];
                    out[(size_t)gr * NOUT + gc] = lin * fabsf(lin) * izn_s[rl] * iw;
                }
            }
        }
    }
}

extern "C" void kernel_launch(void* const* d_in, const int* in_sizes, int n_in,
                              void* d_out, int out_size, void* d_ws, size_t ws_size,
                              hipStream_t stream) {
    const float* z = (const float*)d_in[0];
    const float* w = (const float*)d_in[1];
    float* out     = (float*)d_out;
    float* iwn     = (float*)d_ws;          // 512 floats = 2 KB scratch
    const int M    = in_sizes[0] / KDIM;    // 100000

    bcos_wnorm<<<NOUT, 256, 0, stream>>>(w, iwn);

    dim3 grid(NOUT / BN, (M + BM - 1) / BM);   // (4, 782): col-blocks fastest
    bcos_gemm<<<grid, 256, 0, stream>>>(z, w, iwn, out, M);
}
